// WinBasedAttention_46445776338977
// MI455X (gfx1250) — compile-verified, matrix operand order checked by reading, not run on hardware
//
#include <hip/hip_runtime.h>
#include <hip/hip_bf16.h>

typedef _Float16 v16h __attribute__((ext_vector_type(16)));
typedef _Float16 v8h  __attribute__((ext_vector_type(8)));
typedef float    v8f  __attribute__((ext_vector_type(8)));

#define WSZ   8
#define SSZ   4
#define HEADS 8
#define HD    24
#define CCH   192
#define NTOK  64
#define HIMG  256
#define SCALE 0.20412414523193154f   // 1/sqrt(24)

#define QKV_FRAGS  (36*6)            // 576/16 N-tiles x 192/32 K-tiles
#define PROJ_FRAGS (12*6)            // 192/16 x 192/32
#define QKV_PACK_H  (QKV_FRAGS*512)  // halves
#define PROJ_PACK_H (PROJ_FRAGS*512)

__device__ __forceinline__ v16h mk16(v8h l8, v8h h8) {
    v16h r;
#pragma unroll
    for (int e = 0; e < 8; ++e) { r[e] = l8[e]; r[e + 8] = h8[e]; }
    return r;
}

__device__ __forceinline__ v8f wmma16(v16h a, v16h b, v8f c) {
    return __builtin_amdgcn_wmma_f32_16x16x32_f16(false, a, false, b, (short)0, c, false, false);
}

// ---- pre-pass: convert f32 weights to f16, pre-swizzled into WMMA B-fragment order ----
// packed[frag][lane][e] = W[N = nt*16 + lane%16][K = kt*32 + kmap(lane>=16, e)]
__global__ __launch_bounds__(256) void pack_weights(const float* __restrict__ qkv_w,
                                                    const float* __restrict__ proj_w,
                                                    _Float16* __restrict__ wpack) {
    int idx = blockIdx.x * 256 + threadIdx.x;
    if (idx >= QKV_PACK_H + PROJ_PACK_H) return;
    bool isProj = idx >= QKV_PACK_H;
    int rel  = isProj ? idx - QKV_PACK_H : idx;
    int frag = rel >> 9;
    int lane = (rel >> 4) & 31;
    int e    = rel & 15;
    int nt = frag / 6, kt = frag % 6;
    int n  = nt * 16 + (lane & 15);
    int hi = lane >> 4;
    int k  = kt * 32 + (e & 7) + ((e & 8) ? 16 : 0) + (hi ? 8 : 0);
    float v = isProj ? proj_w[n * CCH + k] : qkv_w[n * CCH + k];
    wpack[idx] = (_Float16)v;
}

// ---- fused shifted-window attention: one block (8 wave32) per 8x8 window ----
__global__ __launch_bounds__(256) void swin_attn(const float* __restrict__ x,
                                                 const float* __restrict__ qkv_b,
                                                 const float* __restrict__ proj_b,
                                                 const float* __restrict__ rpb,
                                                 const _Float16* __restrict__ wqkv,
                                                 const _Float16* __restrict__ wproj,
                                                 float* __restrict__ out) {
    __shared__ __align__(16) char smem[142112];
    _Float16* sX   = (_Float16*)smem;                 // 64 x 200 f16 (window in; later attn-out)
    _Float16* sQKV = (_Float16*)(smem + 25600);       // 64 x 584 f16 (later 64x196 f32 proj-out)
    float*    sS   = (float*)   (smem + 100352);      // 64 x 68  f32 scores
    _Float16* sP   = (_Float16*)(smem + 117760);      // 64 x 72  f16 probs
    _Float16* sVt  = (_Float16*)(smem + 126976);      // 32 x 72  f16 v^T
    float*    sBq  = (float*)   (smem + 131584);      // 576
    float*    sBp  = (float*)   (smem + 133888);      // 192
    float*    sRPB = (float*)   (smem + 134656);      // 225*8
    int*      sRid = (int*)     (smem + 141856);      // 64 region ids

    const int tid  = threadIdx.x;
    const int lane = tid & 31;
    const int wv   = tid >> 5;
    const int lo   = lane & 15;
    const int hi   = lane >> 4;

    const int blk = blockIdx.x;
    const int b   = blk >> 10;
    const int wr  = (blk >> 5) & 31;
    const int wc  = blk & 31;

    // ---- stage 0: stage biases / rpb / region ids; gather shifted window to f16 LDS ----
    for (int i = tid; i < 576; i += 256) sBq[i] = qkv_b[i];
    if (tid < 192) sBp[tid] = proj_b[tid];
    for (int i = tid; i < 1800; i += 256) sRPB[i] = rpb[i];
    if (tid < 64) {
        int i = tid >> 3, j = tid & 7;
        int hs = wr * 8 + i, wsv = wc * 8 + j;
        int rh = hs  < (HIMG - WSZ) ? 0 : (hs  < (HIMG - SSZ) ? 1 : 2);
        int rw = wsv < (HIMG - WSZ) ? 0 : (wsv < (HIMG - SSZ) ? 1 : 2);
        sRid[tid] = rh * 3 + rw;
    }
    const size_t xbase = ((size_t)b * CCH) << 16;
    for (int idx = tid; idx < CCH * NTOK; idx += 256) {
        int c = idx >> 6, t = idx & 63;
        int i = t >> 3, j = t & 7;
        int hh = (wr * 8 + i + SSZ) & 255;
        int ww = (wc * 8 + j + SSZ) & 255;
        float v = x[xbase + ((size_t)c << 16) + (hh << 8) + ww];
        sX[t * 200 + c] = (_Float16)v;
    }
    __syncthreads();

    // ---- stage 1: QKV GEMM (64x192)@(192x576), bias, scale q, f16 -> sQKV ----
    for (int tt = wv; tt < 144; tt += 8) {
        int mt = tt & 3, nt = tt >> 2;
        v8f acc = {};
#pragma unroll
        for (int kt = 0; kt < 6; ++kt) {
            const _Float16* arow = sX + (mt * 16 + lo) * 200 + kt * 32;
            v8h a0 = *(const v8h*)(arow + (hi ? 8 : 0));
            v8h a1 = *(const v8h*)(arow + (hi ? 24 : 16));
            v16h bfr = *(const v16h*)(wqkv + (size_t)((nt * 6 + kt) * 32 + lane) * 16);
            acc = wmma16(mk16(a0, a1), bfr, acc);
        }
        int col  = nt * 16 + lo;
        float bb = sBq[col];
        float sc = (col < CCH) ? SCALE : 1.0f;
#pragma unroll
        for (int r = 0; r < 8; ++r) {
            int m = mt * 16 + r + (hi ? 8 : 0);
            sQKV[m * 584 + col] = (_Float16)((acc[r] + bb) * sc);
        }
    }
    __syncthreads();

    // ---- stage 2: per-head attention ----
    for (int h = 0; h < HEADS; ++h) {
        // v^T (dim-major), zero-padded to 32 rows
        for (int idx = tid; idx < 32 * 64; idx += 256) {
            int d = idx >> 6, t = idx & 63;
            _Float16 v = (_Float16)0.0f;
            if (d < HD) v = sQKV[t * 584 + 384 + h * HD + d];
            sVt[d * 72 + t] = v;
        }
        // scores = q @ k^T (K=24 zero-padded to 32), + rel-pos bias + shift mask
#pragma unroll
        for (int s = 0; s < 2; ++s) {
            int tt = wv * 2 + s;
            int mt = tt >> 2, nt = tt & 3;
            const _Float16* qrow = sQKV + (mt * 16 + lo) * 584 + h * HD;
            const _Float16* krow = sQKV + (nt * 16 + lo) * 584 + CCH + h * HD;
            v8h z = {};
            v8h a0 = *(const v8h*)(qrow + (hi ? 8 : 0));
            v8h a1; if (hi) a1 = z; else a1 = *(const v8h*)(qrow + 16);
            v8h b0 = *(const v8h*)(krow + (hi ? 8 : 0));
            v8h b1; if (hi) b1 = z; else b1 = *(const v8h*)(krow + 16);
            v8f acc = {};
            acc = wmma16(mk16(a0, a1), mk16(b0, b1), acc);
            int n    = nt * 16 + lo;
            int ridN = sRid[n];
            int rN = n >> 3, cN = n & 7;
#pragma unroll
            for (int r = 0; r < 8; ++r) {
                int m  = mt * 16 + r + (hi ? 8 : 0);
                int rM = m >> 3, cM = m & 7;
                int ridx = (rM - rN + 7) * 15 + (cM - cN + 7);
                float mk = (sRid[m] != ridN) ? -100.0f : 0.0f;
                sS[m * 68 + n] = acc[r] + sRPB[ridx * 8 + h] + mk;
            }
        }
        __syncthreads();
        // softmax over rows of 64: 4 lanes per row, shuffle reduction
        {
            int row = tid >> 2, part = tid & 3;
            const float* srow = sS + row * 68 + part * 16;
            float vals[16];
            float mx = -1e30f;
#pragma unroll
            for (int e = 0; e < 16; ++e) { vals[e] = srow[e]; mx = fmaxf(mx, vals[e]); }
            mx = fmaxf(mx, __shfl_xor(mx, 1, 32));
            mx = fmaxf(mx, __shfl_xor(mx, 2, 32));
            float sum = 0.f;
#pragma unroll
            for (int e = 0; e < 16; ++e) { vals[e] = __expf(vals[e] - mx); sum += vals[e]; }
            sum += __shfl_xor(sum, 1, 32);
            sum += __shfl_xor(sum, 2, 32);
            float inv = 1.0f / sum;
            _Float16* prow = sP + row * 72 + part * 16;
#pragma unroll
            for (int e = 0; e < 16; ++e) prow[e] = (_Float16)(vals[e] * inv);
        }
        __syncthreads();
        // out_h = P @ v : 4x2 tiles (N=24 padded to 32), one tile per wave, K=64 in 2 steps
        {
            int mt = wv >> 1, nt = wv & 1;
            v8f acc = {};
#pragma unroll
            for (int kt = 0; kt < 2; ++kt) {
                const _Float16* arow = sP  + (mt * 16 + lo) * 72 + kt * 32;
                const _Float16* brow = sVt + (nt * 16 + lo) * 72 + kt * 32;
                v8h a0 = *(const v8h*)(arow + (hi ? 8 : 0));
                v8h a1 = *(const v8h*)(arow + (hi ? 24 : 16));
                v8h b0 = *(const v8h*)(brow + (hi ? 8 : 0));
                v8h b1 = *(const v8h*)(brow + (hi ? 24 : 16));
                acc = wmma16(mk16(a0, a1), mk16(b0, b1), acc);
            }
            int d = nt * 16 + lo;
            if (d < HD) {
#pragma unroll
                for (int r = 0; r < 8; ++r) {
                    int m = mt * 16 + r + (hi ? 8 : 0);
                    sX[m * 200 + h * HD + d] = (_Float16)acc[r];   // sX reused as attn-out
                }
            }
        }
        __syncthreads();
    }

    // ---- stage 3: proj GEMM (64x192)@(192x192) + bias -> f32 LDS (overlays sQKV) ----
    float* sPO = (float*)sQKV;   // 64 x 196
    for (int tt = wv; tt < 48; tt += 8) {
        int mt = tt & 3, nt = tt >> 2;
        v8f acc = {};
#pragma unroll
        for (int kt = 0; kt < 6; ++kt) {
            const _Float16* arow = sX + (mt * 16 + lo) * 200 + kt * 32;
            v8h a0 = *(const v8h*)(arow + (hi ? 8 : 0));
            v8h a1 = *(const v8h*)(arow + (hi ? 24 : 16));
            v16h bfr = *(const v16h*)(wproj + (size_t)((nt * 6 + kt) * 32 + lane) * 16);
            acc = wmma16(mk16(a0, a1), bfr, acc);
        }
        int col  = nt * 16 + lo;
        float bb = sBp[col];
#pragma unroll
        for (int r = 0; r < 8; ++r) {
            int m = mt * 16 + r + (hi ? 8 : 0);
            sPO[m * 196 + col] = acc[r] + bb;
        }
    }
    __syncthreads();

    // ---- stage 4: reverse-shift scatter + residual (same coords as the gather) ----
    for (int idx = tid; idx < CCH * NTOK; idx += 256) {
        int c = idx >> 6, t = idx & 63;
        int i = t >> 3, j = t & 7;
        int hh = (wr * 8 + i + SSZ) & 255;
        int ww = (wc * 8 + j + SSZ) & 255;
        size_t g = xbase + ((size_t)c << 16) + (hh << 8) + ww;
        out[g] = x[g] + sPO[t * 196 + c];
    }
}

extern "C" void kernel_launch(void* const* d_in, const int* in_sizes, int n_in,
                              void* d_out, int out_size, void* d_ws, size_t ws_size,
                              hipStream_t stream) {
    (void)in_sizes; (void)n_in; (void)out_size; (void)ws_size;
    const float* x      = (const float*)d_in[0];
    const float* qkv_w  = (const float*)d_in[1];
    const float* qkv_b  = (const float*)d_in[2];
    const float* proj_w = (const float*)d_in[3];
    const float* proj_b = (const float*)d_in[4];
    const float* rpb    = (const float*)d_in[5];
    float* out = (float*)d_out;

    _Float16* wpack = (_Float16*)d_ws;
    int total = QKV_PACK_H + PROJ_PACK_H;
    pack_weights<<<(total + 255) / 256, 256, 0, stream>>>(qkv_w, proj_w, wpack);
    swin_attn<<<4096, 256, 0, stream>>>(x, qkv_b, proj_b, rpb,
                                        wpack, wpack + QKV_PACK_H, out);
}
